// MPNNconv_20203526160531
// MI455X (gfx1250) — matrix operand order repaired and checked by reading, not compiled, not run on hardware
//
#include <hip/hip_runtime.h>
#include <cmath>

typedef __bf16 bf16;
typedef __attribute__((ext_vector_type(16))) __bf16 v16bf;
typedef __attribute__((ext_vector_type(8)))  __bf16 v8bf;
typedef __attribute__((ext_vector_type(8)))  float  v8f;

#define NN   12500
#define NE   50000
#define DIN  128
#define DE   12
#define DD   64
#define DEH  32
#define NSTEPS 2

// ---------------- WMMA helpers (CDNA5 bf16, wave32) ----------------
__device__ __forceinline__ v8f wmma_bf16(v16bf a, v16bf b, v8f c) {
  return __builtin_amdgcn_wmma_f32_16x16x32_bf16(false, a, false, b, (short)0, c,
                                                 false, false);
}

// A-frag (16x32 bf16): lane holds row M=lane&15; element i -> K=2*kb+i,
// kb=(lane>>4)*8.  rowk0 = &A[row][k0].
__device__ __forceinline__ v16bf load_a16(const bf16* __restrict__ rowk0, int kb) {
  v8bf lo = *(const v8bf*)(rowk0 + 2 * kb);
  v8bf hi = *(const v8bf*)(rowk0 + 2 * kb + 8);
  v16bf a;
#pragma unroll
  for (int i = 0; i < 8; ++i) { a[i] = lo[i]; a[i + 8] = hi[i]; }
  return a;
}

// B-frag (32x16 bf16): lane holds column N=lane&15, K = kg+i (kg=(lane>>4)*16).
// p points at Bt[N][k0+kg] where Bt is the transposed (column-major-contiguous) B.
__device__ __forceinline__ v16bf load_b16(const bf16* __restrict__ p) {
  v8bf lo = *(const v8bf*)p;
  v8bf hi = *(const v8bf*)(p + 8);
  v16bf b;
#pragma unroll
  for (int i = 0; i < 8; ++i) { b[i] = lo[i]; b[i + 8] = hi[i]; }
  return b;
}

__device__ __forceinline__ float sigmoidf_(float x) { return 1.0f / (1.0f + __expf(-x)); }

__device__ __forceinline__ void atomic_add_f32(float* p, float v) {
  __hip_atomic_fetch_add(p, v, __ATOMIC_RELAXED, __HIP_MEMORY_SCOPE_AGENT);
}

// ---------------- small utility kernels ----------------
__global__ void k_cvt(const float* __restrict__ s, bf16* __restrict__ d, int n) {
  int i = blockIdx.x * blockDim.x + threadIdx.x;
  if (i < n) d[i] = (bf16)s[i];
}

// dst[n*K + k] = src[k*N + n]  (make Bt layout from row-major [K,N] weight)
__global__ void k_cvt_t(const float* __restrict__ s, bf16* __restrict__ d, int K, int Nn) {
  int i = blockIdx.x * blockDim.x + threadIdx.x;
  if (i < K * Nn) {
    int k = i / Nn, n = i - k * Nn;
    d[(size_t)n * K + k] = (bf16)s[i];
  }
}

__global__ void k_zero(float* __restrict__ p, int n) {
  int i = blockIdx.x * blockDim.x + threadIdx.x;
  if (i < n) p[i] = 0.0f;
}

__global__ void k_copy(const float* __restrict__ s, float* __restrict__ d, int n) {
  int i = blockIdx.x * blockDim.x + threadIdx.x;
  if (i < n) d[i] = s[i];
}

// h_e = relu(edge_attr @ w_e1 + b_e1) -> bf16 [E, 32]; K=12 so plain VALU.
__global__ void k_edge_mlp(const float* __restrict__ ea, const float* __restrict__ w1,
                           const float* __restrict__ b1, bf16* __restrict__ he) {
  int i = blockIdx.x * blockDim.x + threadIdx.x;
  if (i >= NE * DEH) return;
  int e = i >> 5, j = i & 31;
  float s = b1[j];
  const float* er = ea + (size_t)e * DE;
#pragma unroll
  for (int t = 0; t < DE; ++t) s += er[t] * w1[t * DEH + j];
  he[i] = (bf16)(s > 0.f ? s : 0.f);
}

// ---------------- node projection: relu(x @ w_proj + b) ----------------
// wave per 16-row tile, all 4 column tiles, K=128 (4 WMMA k-steps)
__global__ __launch_bounds__(256) void k_proj(const bf16* __restrict__ xb,
                                              const bf16* __restrict__ wT,  // [64][128]
                                              const float* __restrict__ bias,
                                              float* __restrict__ hid,
                                              bf16* __restrict__ hb) {
  int wave = threadIdx.x >> 5, lane = threadIdx.x & 31;
  int mt = blockIdx.x * 8 + wave;
  int mtiles = (NN + 15) >> 4;
  if (mt >= mtiles) return;
  int m0 = mt << 4;
  int l16 = lane & 15, hiW = lane >> 4, kb = hiW * 8, kg = hiW * 16;
  int ar = m0 + l16; if (ar >= NN) ar = NN - 1;
  const bf16* abase = xb + (size_t)ar * DIN;
  v8f acc[4] = {};
#pragma unroll
  for (int k0 = 0; k0 < DIN; k0 += 32) {
    v16bf a = load_a16(abase + k0, kb);
#pragma unroll
    for (int nt = 0; nt < 4; ++nt) {
      const bf16* bp = wT + (size_t)(nt * 16 + l16) * DIN + k0 + kg;
      acc[nt] = wmma_bf16(a, load_b16(bp), acc[nt]);
    }
  }
#pragma unroll
  for (int nt = 0; nt < 4; ++nt) {
    int col = nt * 16 + l16;
    float bb = bias[col];
#pragma unroll
    for (int r = 0; r < 8; ++r) {
      int m = m0 + hiW * 8 + r;                 // C layout: M = r + 8*(lane>>4)
      if (m < NN) {
        float v = acc[nt][r] + bb;
        v = v > 0.f ? v : 0.f;
        hid[(size_t)m * DD + col] = v;
        hb[(size_t)m * DD + col] = (bf16)v;
      }
    }
  }
}

// ---------------- fused NNConv message GEMM + scatter ----------------
// msg = U[E,2112] @ Tt, U[e, k*64+d] = h_e[k]*node[src_e][d] (last 64 rows: h=1 -> b_e2)
// Tt = [64 cols][2112 K] bf16, resident in 320KB LDS (CDNA5 WGP LDS), padded stride.
constexpr int TT_STRIDE = 2120;               // 2112 + 8 halves pad (16B aligned, bank spread)
constexpr int KCHUNKS   = 66;                 // 64 chunks from w_e2 + 2 from b_e2
constexpr int MSG_WAVES = 12;
constexpr int NODE_OFF  = 64 * TT_STRIDE;                   // halves
constexpr int H_OFF     = NODE_OFF + MSG_WAVES * 16 * 64;   // halves
constexpr int SMEM_HALVES = H_OFF + MSG_WAVES * 16 * 32;
constexpr size_t MSG_SMEM_BYTES = (size_t)SMEM_HALVES * 2;  // 308,224 B <= 320KB WGP LDS

__global__ __launch_bounds__(MSG_WAVES * 32) void k_msg(
    const bf16* __restrict__ hb, const bf16* __restrict__ he,
    const float* __restrict__ w_e2, const float* __restrict__ b_e2,
    const int* __restrict__ eidx, float* __restrict__ agg) {
  extern __shared__ __align__(16) char smem_raw[];
  bf16* sm = (bf16*)smem_raw;
  bf16* tt = sm;

  // One-time fill: w_e2 viewed as [2048,64] row-major (r = k*64+d), transposed into LDS.
  for (int idx = threadIdx.x; idx < 2048 * 64; idx += blockDim.x) {
    int r = idx >> 6, n = idx & 63;
    tt[n * TT_STRIDE + r] = (bf16)w_e2[idx];
  }
  // b_e2 reshaped [64,64]: extra K rows with implicit h=1
  for (int idx = threadIdx.x; idx < 64 * 64; idx += blockDim.x) {
    int d = idx >> 6, n = idx & 63;
    tt[n * TT_STRIDE + 2048 + d] = (bf16)b_e2[idx];
  }
  __syncthreads();

  const int wave = threadIdx.x >> 5, lane = threadIdx.x & 31;
  const int l16 = lane & 15, hiW = lane >> 4;
  const int kb = hiW * 8, kg = hiW * 16;
  bf16* nstage = sm + NODE_OFF + wave * (16 * 64);
  bf16* hstage = sm + H_OFF + wave * (16 * 32);
  const bf16* arow = nstage + l16 * 64;   // A row for this lane = edge l16
  const bf16* hrow = hstage + l16 * 32;

  for (int t = blockIdx.x * MSG_WAVES + wave; t < NE / 16; t += gridDim.x * MSG_WAVES) {
    {  // stage 16 src-node rows (16x64 bf16) and 16 h rows (16x32 bf16) per wave
      int e = t * 16 + l16;
      int s = eidx[e];  // edge_index[0][e] = src
      const v8bf* gn = (const v8bf*)(hb + (size_t)s * DD + hiW * 32);
      v8bf* ln = (v8bf*)(nstage + l16 * 64 + hiW * 32);
      ln[0] = gn[0]; ln[1] = gn[1]; ln[2] = gn[2]; ln[3] = gn[3];
      const v8bf* gh = (const v8bf*)(he + (size_t)e * DEH + hiW * 16);
      v8bf* lh = (v8bf*)(hstage + l16 * 32 + hiW * 16);
      lh[0] = gh[0]; lh[1] = gh[1];
    }
    __asm__ volatile("s_wait_dscnt 0" ::: "memory");  // LDS in-order; pin compiler too

    v8f a0 = {}, a1 = {}, a2 = {}, a3 = {};
    for (int j = 0; j < KCHUNKS; ++j) {
      const int d0 = (j & 1) << 5;                         // d window within node row
      const float hv = (j < 64) ? (float)hrow[j >> 1] : 1.0f;
      v8bf nlo = *(const v8bf*)(arow + d0 + 2 * kb);
      v8bf nhi = *(const v8bf*)(arow + d0 + 2 * kb + 8);
      v16bf afrag;                                          // rank-1 U row chunk
#pragma unroll
      for (int i = 0; i < 8; ++i) {
        afrag[i]     = (bf16)(hv * (float)nlo[i]);
        afrag[i + 8] = (bf16)(hv * (float)nhi[i]);
      }
      const bf16* tb = tt + l16 * TT_STRIDE + j * 32 + kg;
      a0 = wmma_bf16(afrag, load_b16(tb),                  a0);
      a1 = wmma_bf16(afrag, load_b16(tb + 16 * TT_STRIDE), a1);
      a2 = wmma_bf16(afrag, load_b16(tb + 32 * TT_STRIDE), a2);
      a3 = wmma_bf16(afrag, load_b16(tb + 48 * TT_STRIDE), a3);
    }
    // scatter-add msg tile [16 edges x 64] into agg[dst]
#pragma unroll
    for (int r = 0; r < 8; ++r) {
      int e = t * 16 + hiW * 8 + r;          // C layout: M = r + 8*(lane>>4)
      int dn = eidx[NE + e];                  // edge_index[1][e] = dst
      float* ap = agg + (size_t)dn * DD + l16;
      atomic_add_f32(ap +  0, a0[r]);
      atomic_add_f32(ap + 16, a1[r]);
      atomic_add_f32(ap + 32, a2[r]);
      atomic_add_f32(ap + 48, a3[r]);
    }
  }
}

// ---------------- conv combine: relu(agg + node @ w_root + b_conv) ----------------
__global__ __launch_bounds__(256) void k_conv(const bf16* __restrict__ nodeb,
                                              const bf16* __restrict__ wrootT,  // [64][64]
                                              const float* __restrict__ agg,
                                              const float* __restrict__ bconv,
                                              bf16* __restrict__ convb) {
  int wave = threadIdx.x >> 5, lane = threadIdx.x & 31;
  int mt = blockIdx.x * 8 + wave;
  int mtiles = (NN + 15) >> 4;
  if (mt >= mtiles) return;
  int m0 = mt << 4;
  int l16 = lane & 15, hiW = lane >> 4, kb = hiW * 8, kg = hiW * 16;
  int ar = m0 + l16; if (ar >= NN) ar = NN - 1;
  const bf16* abase = nodeb + (size_t)ar * DD;
  v16bf af0 = load_a16(abase, kb);
  v16bf af1 = load_a16(abase + 32, kb);
  v8f acc[4] = {};
#pragma unroll
  for (int nt = 0; nt < 4; ++nt) {
    const bf16* bp = wrootT + (size_t)(nt * 16 + l16) * DD;
    acc[nt] = wmma_bf16(af0, load_b16(bp + kg), acc[nt]);
    acc[nt] = wmma_bf16(af1, load_b16(bp + 32 + kg), acc[nt]);
  }
#pragma unroll
  for (int nt = 0; nt < 4; ++nt) {
    int col = nt * 16 + l16;
    float bb = bconv[col];
#pragma unroll
    for (int r = 0; r < 8; ++r) {
      int m = m0 + hiW * 8 + r;
      if (m < NN) {
        float v = acc[nt][r] + agg[(size_t)m * DD + col] + bb;
        convb[(size_t)m * DD + col] = (bf16)(v > 0.f ? v : 0.f);
      }
    }
  }
}

// ---------------- fused GRU step ----------------
// gi = node @ w_ih.T, gh = hidden @ w_hh.T; w_ih row-major [192,64] is already Bt layout.
__global__ __launch_bounds__(256) void k_gru(const bf16* __restrict__ convb,
                                             const bf16* __restrict__ hb,
                                             const bf16* __restrict__ wihb,  // [192][64]
                                             const bf16* __restrict__ whhb,  // [192][64]
                                             const float* __restrict__ b_ih,
                                             const float* __restrict__ b_hh,
                                             float* __restrict__ hid,
                                             bf16* __restrict__ hb_out) {
  int wave = threadIdx.x >> 5, lane = threadIdx.x & 31;
  int mt = blockIdx.x * 8 + wave;
  int mtiles = (NN + 15) >> 4;
  if (mt >= mtiles) return;
  int m0 = mt << 4;
  int l16 = lane & 15, hiW = lane >> 4, kb = hiW * 8, kg = hiW * 16;
  int ar = m0 + l16; if (ar >= NN) ar = NN - 1;
  // preload all A fragments (hb is also our output; read-before-write within wave)
  const bf16* xr = convb + (size_t)ar * DD;
  const bf16* hr = hb + (size_t)ar * DD;
  v16bf ax0 = load_a16(xr, kb),      ax1 = load_a16(xr + 32, kb);
  v16bf ah0 = load_a16(hr, kb),      ah1 = load_a16(hr + 32, kb);

#pragma unroll
  for (int nt = 0; nt < 4; ++nt) {
    int colb = nt * 16 + l16;
    v8f ir = {}, iz = {}, inn = {}, hrr = {}, hz = {}, hn = {};
#pragma unroll
    for (int k0 = 0; k0 < DD; k0 += 32) {
      v16bf ax = (k0 == 0) ? ax0 : ax1;
      v16bf ah = (k0 == 0) ? ah0 : ah1;
      ir  = wmma_bf16(ax, load_b16(wihb + (size_t)(colb)       * DD + k0 + kg), ir);
      iz  = wmma_bf16(ax, load_b16(wihb + (size_t)(64 + colb)  * DD + k0 + kg), iz);
      inn = wmma_bf16(ax, load_b16(wihb + (size_t)(128 + colb) * DD + k0 + kg), inn);
      hrr = wmma_bf16(ah, load_b16(whhb + (size_t)(colb)       * DD + k0 + kg), hrr);
      hz  = wmma_bf16(ah, load_b16(whhb + (size_t)(64 + colb)  * DD + k0 + kg), hz);
      hn  = wmma_bf16(ah, load_b16(whhb + (size_t)(128 + colb) * DD + k0 + kg), hn);
    }
    float bir = b_ih[colb], biz = b_ih[64 + colb], bin = b_ih[128 + colb];
    float bhr = b_hh[colb], bhz = b_hh[64 + colb], bhn = b_hh[128 + colb];
#pragma unroll
    for (int r = 0; r < 8; ++r) {
      int m = m0 + hiW * 8 + r;
      if (m < NN) {
        float rg = sigmoidf_((ir[r] + bir) + (hrr[r] + bhr));
        float zg = sigmoidf_((iz[r] + biz) + (hz[r] + bhz));
        float ng = tanhf((inn[r] + bin) + rg * (hn[r] + bhn));
        size_t o = (size_t)m * DD + colb;
        float hold = hid[o];
        float hnew = (1.0f - zg) * ng + zg * hold;
        hid[o] = hnew;
        hb_out[o] = (bf16)hnew;
      }
    }
  }
}

// ---------------- launch ----------------
extern "C" void kernel_launch(void* const* d_in, const int* in_sizes, int n_in,
                              void* d_out, int out_size, void* d_ws, size_t ws_size,
                              hipStream_t stream) {
  (void)in_sizes; (void)n_in; (void)ws_size;
  const float* x      = (const float*)d_in[0];
  const int*   eidx   = (const int*)d_in[1];    // [2, E] (src row then dst row)
  const float* eattr  = (const float*)d_in[2];
  const float* w_proj = (const float*)d_in[3];
  const float* b_proj = (const float*)d_in[4];
  const float* w_e1   = (const float*)d_in[5];
  const float* b_e1   = (const float*)d_in[6];
  const float* w_e2   = (const float*)d_in[7];
  const float* b_e2   = (const float*)d_in[8];
  const float* w_root = (const float*)d_in[9];
  const float* b_conv = (const float*)d_in[10];
  const float* w_ih   = (const float*)d_in[11];
  const float* w_hh   = (const float*)d_in[12];
  const float* b_ih   = (const float*)d_in[13];
  const float* b_hh   = (const float*)d_in[14];

  char* ws = (char*)d_ws;
  size_t off = 0;
  auto alloc = [&](size_t bytes) -> char* {
    char* p = ws + off; off += (bytes + 511) & ~(size_t)511; return p;
  };
  float* hid    = (float*)alloc((size_t)NN * DD * 4);
  float* agg    = (float*)alloc((size_t)NN * DD * 4);
  bf16*  xb     = (bf16*)alloc((size_t)NN * DIN * 2);
  bf16*  hb     = (bf16*)alloc((size_t)NN * DD * 2);
  bf16*  convb  = (bf16*)alloc((size_t)NN * DD * 2);
  bf16*  he     = (bf16*)alloc((size_t)NE * DEH * 2);
  bf16*  wprojT = (bf16*)alloc((size_t)DIN * DD * 2);
  bf16*  wrootT = (bf16*)alloc((size_t)DD * DD * 2);
  bf16*  wihb   = (bf16*)alloc((size_t)3 * DD * DD * 2);
  bf16*  whhb   = (bf16*)alloc((size_t)3 * DD * DD * 2);

  const int T = 256;
  auto blk = [](int n, int t) { return (n + t - 1) / t; };

  // weight prep (bf16 copies; transposed where strided B access would result)
  k_cvt<<<blk(NN * DIN, T), T, 0, stream>>>(x, xb, NN * DIN);
  k_cvt<<<blk(3 * DD * DD, T), T, 0, stream>>>(w_ih, wihb, 3 * DD * DD);
  k_cvt<<<blk(3 * DD * DD, T), T, 0, stream>>>(w_hh, whhb, 3 * DD * DD);
  k_cvt_t<<<blk(DIN * DD, T), T, 0, stream>>>(w_proj, wprojT, DIN, DD);
  k_cvt_t<<<blk(DD * DD, T), T, 0, stream>>>(w_root, wrootT, DD, DD);

  // edge MLP hidden
  k_edge_mlp<<<blk(NE * DEH, T), T, 0, stream>>>(eattr, w_e1, b_e1, he);

  // node projection -> hid (f32 master) + hb (bf16 shadow)
  const int mtiles = (NN + 15) / 16;
  const int gemmBlocks = (mtiles + 7) / 8;
  k_proj<<<gemmBlocks, 256, 0, stream>>>(xb, wprojT, b_proj, hid, hb);

  for (int step = 0; step < NSTEPS; ++step) {
    k_zero<<<blk(NN * DD, T), T, 0, stream>>>(agg, NN * DD);
    k_msg<<<96, MSG_WAVES * 32, MSG_SMEM_BYTES, stream>>>(hb, he, w_e2, b_e2, eidx, agg);
    k_conv<<<gemmBlocks, 256, 0, stream>>>(hb, wrootT, agg, b_conv, convb);
    k_gru<<<gemmBlocks, 256, 0, stream>>>(convb, hb, wihb, whhb, b_ih, b_hh, hid, hb);
  }

  k_copy<<<blk(out_size, T), T, 0, stream>>>(hid, (float*)d_out, out_size);
}
